// GATAggregator_13941463843330
// MI455X (gfx1250) — matrix operand order repaired and checked by reading, not compiled, not run on hardware
//
#include <hip/hip_runtime.h>
#include <cstdint>
#include <cstddef>

// ---------------------------------------------------------------------------
// GAT aggregator for MI455X (gfx1250, wave32, WMMA).
//  * cov-path GEMMs (K=256) run on v_wmma_f32_16x16x32_bf16 with pre-swizzled
//    bf16 fragments (contiguous 32B/lane loads).
//  * edge softmax/aggregation uses a dst-CSR built on device (no fp atomics).
//  * xh_cov kept in bf16 (102 MB) so the random edge gather stays L2-resident.
// ---------------------------------------------------------------------------

typedef __attribute__((ext_vector_type(16))) __bf16 v16bf;
typedef __attribute__((ext_vector_type(8)))  float  v8f;

__device__ __forceinline__ unsigned short f2bf(float x) {
  union { float f; unsigned u; } v; v.f = x;
  unsigned u = v.u;
  return (unsigned short)((u + 0x7FFFu + ((u >> 16) & 1u)) >> 16);   // RNE
}
__device__ __forceinline__ float bf2f(unsigned short h) {
  union { unsigned u; float f; } v; v.u = ((unsigned)h) << 16;
  return v.f;
}
__device__ __forceinline__ float lrelu(float a) { return a > 0.f ? a : 0.2f * a; }

// ============================ CSR construction =============================

__global__ void k_deg_init(int* __restrict__ deg, int n) {
  int i = blockIdx.x * blockDim.x + threadIdx.x;
  if (i < n) deg[i] = 1;                       // self loop
}

__global__ void k_deg_count(const int* __restrict__ dst, int* __restrict__ deg, int e) {
  int i = blockIdx.x * blockDim.x + threadIdx.x;
  if (i < e) atomicAdd(&deg[dst[i]], 1);
}

// single 1024-thread block exclusive scan over deg -> rowptr[0..n], rowptr[n]=total
__global__ void k_scan(const int* __restrict__ deg, int* __restrict__ rowptr, int n) {
  __shared__ int buf[1024];
  __shared__ int carry;
  int tid = threadIdx.x;
  if (tid == 0) carry = 0;
  __syncthreads();
  for (int base = 0; base < n; base += 1024) {
    int v = (base + tid < n) ? deg[base + tid] : 0;
    buf[tid] = v;
    __syncthreads();
    for (int off = 1; off < 1024; off <<= 1) {
      int t = (tid >= off) ? buf[tid - off] : 0;
      __syncthreads();
      buf[tid] += t;
      __syncthreads();
    }
    if (base + tid < n) rowptr[base + tid] = carry + buf[tid] - v;
    __syncthreads();
    if (tid == 0) carry += buf[1023];
    __syncthreads();
  }
  if (tid == 0) rowptr[n] = carry;
}

__global__ void k_cursor(const int* __restrict__ rowptr, int* __restrict__ cur, int n) {
  int i = blockIdx.x * blockDim.x + threadIdx.x;
  if (i < n) cur[i] = rowptr[i];
}

__global__ void k_scatter(const int* __restrict__ src, const int* __restrict__ dst,
                          int* __restrict__ cur, int* __restrict__ srcs, int e, int n) {
  int i = blockIdx.x * blockDim.x + threadIdx.x;
  if (i >= e + n) return;
  int s, d;
  if (i < e) { s = src[i]; d = dst[i]; } else { s = d = i - e; }
  int pos = atomicAdd(&cur[d], 1);
  srcs[pos] = s;
}

// ====================== bf16 WMMA fragment swizzles ========================
// A (16-bit, 16x32):  lane = 16*khalf + m ; element e -> VGPR e/2, pos e%2 ;
//   K = (e>=8?16:0) + ((e/2)&3)*2 + (e&1) + khalf*8
// B (16-bit, 32x16):  lane = 16*khalf + ncol ; element e -> K = (e>=8?16:0) + (e&7) + khalf*8

__global__ void k_swizzle_A(const float* __restrict__ X, unsigned short* __restrict__ Af,
                            int tilesM, int K) {
  int ktiles = K >> 5;
  int idx = blockIdx.x * blockDim.x + threadIdx.x;
  if (idx >= tilesM * ktiles * 32) return;
  int lane  = idx & 31;
  int k0    = (idx >> 5) % ktiles;
  int tm    = (idx >> 5) / ktiles;
  int m     = lane & 15, khalf = lane >> 4;
  const float* row = X + (size_t)(tm * 16 + m) * K + k0 * 32;
  unsigned short* dst = Af + (size_t)idx * 16;
#pragma unroll
  for (int e = 0; e < 16; ++e) {
    int vg = e >> 1, pos = e & 1;
    int kl = ((vg >= 4) ? 16 : 0) + ((vg & 3) << 1) + pos + (khalf << 3);
    dst[e] = f2bf(row[kl]);
  }
}

__global__ void k_swizzle_W(const float* __restrict__ W, unsigned short* __restrict__ Wf,
                            int tilesN, int ktiles, int ldw) {
  int idx = blockIdx.x * blockDim.x + threadIdx.x;
  if (idx >= tilesN * ktiles * 32) return;
  int lane  = idx & 31;
  int k0    = (idx >> 5) % ktiles;
  int tn    = (idx >> 5) / ktiles;
  int col   = tn * 16 + (lane & 15);
  int krow0 = k0 * 32 + (lane >> 4) * 8;
  unsigned short* dst = Wf + (size_t)idx * 16;
#pragma unroll
  for (int e = 0; e < 16; ++e)
    dst[e] = f2bf(W[(size_t)(krow0 + e + ((e >= 8) ? 8 : 0)) * ldw + col]);
}

// ============================== WMMA GEMM ==================================
// One wave computes one 16x16 tile of out = A[M,K] @ W[K,N] (+bias).
// Fragments are pre-swizzled: one contiguous 32B load per operand per k-step.

__global__ void k_wmma_gemm(const unsigned short* __restrict__ Af,
                            const unsigned short* __restrict__ Wf,
                            const float* __restrict__ bias,
                            float* __restrict__ outF,
                            unsigned short* __restrict__ outBF,
                            int tilesM, int tilesN, int ldo, int ktiles) {
  int lane = threadIdx.x & 31;
  int gw = blockIdx.x * (blockDim.x >> 5) + (threadIdx.x >> 5);   // wave-uniform
  if (gw >= tilesM * tilesN) return;                              // EXEC stays all-1
  int tm = gw / tilesN, tn = gw % tilesN;

  const v16bf* Ap = (const v16bf*)Af + (size_t)tm * ktiles * 32 + lane;
  const v16bf* Bp = (const v16bf*)Wf + (size_t)tn * ktiles * 32 + lane;

  v8f acc = {};
#pragma unroll
  for (int k = 0; k < 8; ++k) {                                   // ktiles == 8 here
    v16bf a = Ap[k * 32];
    v16bf b = Bp[k * 32];
    acc = __builtin_amdgcn_wmma_f32_16x16x32_bf16(false, a, false, b,
                                                  (short)0, acc, false, false);
  }
  (void)ktiles;

  int col   = tn * 16 + (lane & 15);
  int rbase = tm * 16 + ((lane >> 4) << 3);
  float bv  = bias ? bias[col] : 0.f;
#pragma unroll
  for (int r = 0; r < 8; ++r) {
    float v = acc[r] + bv;
    size_t o = (size_t)(rbase + r) * ldo + col;
    if (outBF) outBF[o] = f2bf(v);
    else       outF[o]  = v;
  }
}

// =========================== attention logits ==============================

__global__ void k_att_cov(const unsigned short* __restrict__ xh,
                          const float* __restrict__ att_s, const float* __restrict__ att_d,
                          float* __restrict__ a_s, float* __restrict__ a_d, int n_nodes) {
  int lane = threadIdx.x & 31;
  int w = blockIdx.x * (blockDim.x >> 5) + (threadIdx.x >> 5);
  if (w >= n_nodes * 4) return;
  int n = w >> 2, h = w & 3;
  const unsigned short* xr = xh + (size_t)n * 1024 + h * 256;
  const float* as = att_s + h * 256;
  const float* ad = att_d + h * 256;
  float s = 0.f, d = 0.f;
#pragma unroll
  for (int k = 0; k < 8; ++k) {
    float x = bf2f(xr[lane + k * 32]);
    s += x * as[lane + k * 32];
    d += x * ad[lane + k * 32];
  }
  for (int off = 16; off; off >>= 1) {
    s += __shfl_xor(s, off, 32);
    d += __shfl_xor(d, off, 32);
  }
  if (lane == 0) { a_s[w] = s; a_d[w] = d; }
}

__global__ void k_mean_xh(const float* __restrict__ x, const float* __restrict__ W,
                          float* __restrict__ xh, int n_nodes) {
  int idx = blockIdx.x * blockDim.x + threadIdx.x;
  if (idx >= n_nodes * 64) return;
  int n = idx >> 6, j = idx & 63;
  const float* row = x + (size_t)n * 16;
  float s = 0.f;
#pragma unroll
  for (int i = 0; i < 16; ++i) s += row[i] * W[i * 64 + j];
  xh[idx] = s;
}

__global__ void k_att_mean(const float* __restrict__ xh,
                           const float* __restrict__ att_s, const float* __restrict__ att_d,
                           float* __restrict__ a_s, float* __restrict__ a_d, int n_nodes) {
  int w = blockIdx.x * blockDim.x + threadIdx.x;
  if (w >= n_nodes * 4) return;
  int n = w >> 2, h = w & 3;
  const float* xr = xh + (size_t)n * 64 + h * 16;
  float s = 0.f, d = 0.f;
#pragma unroll
  for (int c = 0; c < 16; ++c) { s += xr[c] * att_s[h * 16 + c]; d += xr[c] * att_d[h * 16 + c]; }
  a_s[w] = s; a_d[w] = d;
}

// ===================== per-dst online softmax (one wave/node) ==============

__global__ void k_softmax_stats(const int* __restrict__ rowptr, const int* __restrict__ srcs,
                                const float* __restrict__ a_src, const float* __restrict__ a_dst,
                                float* __restrict__ amax, float* __restrict__ denom, int n_nodes) {
  int lane = threadIdx.x & 31;
  int n = blockIdx.x * (blockDim.x >> 5) + (threadIdx.x >> 5);
  if (n >= n_nodes) return;
  int beg = rowptr[n], end = rowptr[n + 1];
  float ad[4], m[4], s[4];
#pragma unroll
  for (int h = 0; h < 4; ++h) { ad[h] = a_dst[n * 4 + h]; m[h] = -INFINITY; s[h] = 0.f; }
  for (int i = beg + lane; i < end; i += 32) {
    int sr = srcs[i];
#pragma unroll
    for (int h = 0; h < 4; ++h) {
      float a = lrelu(a_src[sr * 4 + h] + ad[h]);
      if (a > m[h]) { s[h] = s[h] * __expf(m[h] - a) + 1.f; m[h] = a; }
      else          { s[h] += __expf(a - m[h]); }
    }
  }
  for (int off = 16; off; off >>= 1) {
#pragma unroll
    for (int h = 0; h < 4; ++h) {
      float mo = __shfl_xor(m[h], off, 32);
      float so = __shfl_xor(s[h], off, 32);
      float M  = fmaxf(m[h], mo);
      float c1 = (m[h] == -INFINITY) ? 0.f : s[h] * __expf(m[h] - M);
      float c2 = (mo   == -INFINITY) ? 0.f : so   * __expf(mo   - M);
      m[h] = M; s[h] = c1 + c2;
    }
  }
  if (lane == 0) {
#pragma unroll
    for (int h = 0; h < 4; ++h) { amax[n * 4 + h] = m[h]; denom[n * 4 + h] = s[h]; }
  }
}

// ============== aggregation: one 256-thread block per node (cov) ===========

__global__ void k_aggregate_cov(const int* __restrict__ rowptr, const int* __restrict__ srcs,
                                const float* __restrict__ a_src, const float* __restrict__ a_dst,
                                const float* __restrict__ amax, const float* __restrict__ denom,
                                const unsigned short* __restrict__ xh,
                                const float* __restrict__ bias, float* __restrict__ out) {
  __shared__ int   s_src[64];
  __shared__ float s_coef[64 * 4];
  int n = blockIdx.x, tid = threadIdx.x;
  int beg = rowptr[n], end = rowptr[n + 1];
  float acc = 0.f;
  for (int chunk = beg; chunk < end; chunk += 64) {
    int cnt = min(64, end - chunk);
    __syncthreads();
    if (tid < cnt) {
      int sr = srcs[chunk + tid];
      s_src[tid] = sr;
#pragma unroll
      for (int h = 0; h < 4; ++h) {
        float a = lrelu(a_src[sr * 4 + h] + a_dst[n * 4 + h]);
        s_coef[tid * 4 + h] = __expf(a - amax[n * 4 + h]) / (denom[n * 4 + h] + 1e-16f);
      }
    }
    __syncthreads();
    for (int e = 0; e < cnt; ++e) {
      const unsigned short* xr = xh + (size_t)s_src[e] * 1024 + tid;   // coalesced 512B/(e,h)
#pragma unroll
      for (int h = 0; h < 4; ++h)
        acc += s_coef[e * 4 + h] * bf2f(xr[h * 256]);
    }
  }
  out[(size_t)n * 256 + tid] = acc * 0.25f + bias[tid];
}

// 4 nodes per 256-thread block (mean, C=16, H=4)
__global__ void k_aggregate_mean(const int* __restrict__ rowptr, const int* __restrict__ srcs,
                                 const float* __restrict__ a_src, const float* __restrict__ a_dst,
                                 const float* __restrict__ amax, const float* __restrict__ denom,
                                 const float* __restrict__ xh, const float* __restrict__ bias,
                                 float* __restrict__ out, int n_nodes) {
  __shared__ float red[256];
  int tid = threadIdx.x;
  int local = tid & 63;
  int n = blockIdx.x * 4 + (tid >> 6);
  int h = local >> 4, c = local & 15;
  float acc = 0.f;
  if (n < n_nodes) {
    int beg = rowptr[n], end = rowptr[n + 1];
    float ad = a_dst[n * 4 + h], mx = amax[n * 4 + h], dn = denom[n * 4 + h] + 1e-16f;
    for (int i = beg; i < end; ++i) {
      int sr = srcs[i];
      float a = lrelu(a_src[sr * 4 + h] + ad);
      acc += (__expf(a - mx) / dn) * xh[(size_t)sr * 64 + h * 16 + c];
    }
  }
  red[tid] = acc;
  __syncthreads();
  if (local < 16 && n < n_nodes) {
    int b = tid & ~63;
    float s = red[b + c] + red[b + 16 + c] + red[b + 32 + c] + red[b + 48 + c];
    out[(size_t)n * 16 + c] = s * 0.25f + bias[c];
  }
}

// =========================== tiny 16x16 linears ============================

__global__ void k_fc16(const float* __restrict__ in, const float* __restrict__ W,
                       const float* __restrict__ b, float* __restrict__ out, int n_nodes) {
  int idx = blockIdx.x * blockDim.x + threadIdx.x;
  if (idx >= n_nodes * 16) return;
  int n = idx >> 4, c = idx & 15;
  const float* row = in + (size_t)n * 16;
  float s = b[c];
#pragma unroll
  for (int i = 0; i < 16; ++i) s += row[i] * W[i * 16 + c];
  out[idx] = s;
}

// ================================ launch ===================================

static inline int cdiv(int a, int b) { return (a + b - 1) / b; }

extern "C" void kernel_launch(void* const* d_in, const int* in_sizes, int n_in,
                              void* d_out, int out_size, void* d_ws, size_t ws_size,
                              hipStream_t stream) {
  const int N = in_sizes[0] / 16;          // 50000
  const int E = in_sizes[2] / 2;           // 800000
  const int tilesM = N / 16;               // 3125 (N is a multiple of 16)

  const float* mean     = (const float*)d_in[0];
  const float* cov      = (const float*)d_in[1];
  const int*   ei       = (const int*)d_in[2];
  const int*   e_src    = ei;
  const int*   e_dst    = ei + E;
  const float* W_mean   = (const float*)d_in[3];
  const float* as_mean  = (const float*)d_in[4];
  const float* ad_mean  = (const float*)d_in[5];
  const float* b_mean   = (const float*)d_in[6];
  const float* W_cov    = (const float*)d_in[7];
  const float* as_cov   = (const float*)d_in[8];
  const float* ad_cov   = (const float*)d_in[9];
  const float* b_cov    = (const float*)d_in[10];
  const float* fm_W1    = (const float*)d_in[11];
  const float* fm_b1    = (const float*)d_in[12];
  const float* fm_W2    = (const float*)d_in[13];
  const float* fm_b2    = (const float*)d_in[14];
  const float* fcv_W1   = (const float*)d_in[15];
  const float* fcv_b1   = (const float*)d_in[16];
  const float* fcv_W2   = (const float*)d_in[17];
  const float* fcv_b2   = (const float*)d_in[18];

  float* mean_out = (float*)d_out;
  float* cov_out  = (float*)d_out + (size_t)N * 16;

  // -------- workspace layout (sequential, 256B aligned) --------
  char* wp = (char*)d_ws;
  auto alloc = [&](size_t bytes) -> void* {
    void* p = wp; wp += (bytes + 255) & ~(size_t)255; return p;
  };
  int*   deg      = (int*)alloc((size_t)N * 4);
  int*   rowptr   = (int*)alloc((size_t)(N + 1) * 4);
  int*   cursor   = (int*)alloc((size_t)N * 4);
  int*   srcs     = (int*)alloc((size_t)(E + N) * 4);
  float* a_src_m  = (float*)alloc((size_t)N * 4 * 4);
  float* a_dst_m  = (float*)alloc((size_t)N * 4 * 4);
  float* amax_m   = (float*)alloc((size_t)N * 4 * 4);
  float* den_m    = (float*)alloc((size_t)N * 4 * 4);
  float* a_src_c  = (float*)alloc((size_t)N * 4 * 4);
  float* a_dst_c  = (float*)alloc((size_t)N * 4 * 4);
  float* amax_c   = (float*)alloc((size_t)N * 4 * 4);
  float* den_c    = (float*)alloc((size_t)N * 4 * 4);
  float* xh_mean  = (float*)alloc((size_t)N * 64 * 4);
  unsigned short* xh_cov = (unsigned short*)alloc((size_t)N * 1024 * 2);   // bf16, L2-resident
  float* mean_agg = (float*)alloc((size_t)N * 16 * 4);
  float* cov_agg  = (float*)alloc((size_t)N * 256 * 4);
  float* t_mean   = (float*)alloc((size_t)N * 16 * 4);
  float* t_cov    = (float*)alloc((size_t)N * 256 * 4);
  unsigned short* Afrag  = (unsigned short*)alloc((size_t)tilesM * 8 * 32 * 16 * 2);
  unsigned short* WfragC = (unsigned short*)alloc((size_t)64 * 8 * 32 * 16 * 2);
  unsigned short* Wfrag1 = (unsigned short*)alloc((size_t)16 * 8 * 32 * 16 * 2);
  unsigned short* Wfrag2 = (unsigned short*)alloc((size_t)16 * 8 * 32 * 16 * 2);
  (void)ws_size; (void)n_in; (void)out_size;

  // -------- CSR by destination (shared by both GAT layers) --------
  k_deg_init <<<cdiv(N, 256), 256, 0, stream>>>(deg, N);
  k_deg_count<<<cdiv(E, 256), 256, 0, stream>>>(e_dst, deg, E);
  k_scan     <<<1, 1024, 0, stream>>>(deg, rowptr, N);
  k_cursor   <<<cdiv(N, 256), 256, 0, stream>>>(rowptr, cursor, N);
  k_scatter  <<<cdiv(E + N, 256), 256, 0, stream>>>(e_src, e_dst, cursor, srcs, E, N);

  // -------- mean path (tiny; scalar kernels) --------
  k_mean_xh       <<<cdiv(N * 64, 256), 256, 0, stream>>>(mean, W_mean, xh_mean, N);
  k_att_mean      <<<cdiv(N * 4, 256), 256, 0, stream>>>(xh_mean, as_mean, ad_mean,
                                                         a_src_m, a_dst_m, N);
  k_softmax_stats <<<cdiv(N, 8), 256, 0, stream>>>(rowptr, srcs, a_src_m, a_dst_m,
                                                   amax_m, den_m, N);
  k_aggregate_mean<<<cdiv(N, 4), 256, 0, stream>>>(rowptr, srcs, a_src_m, a_dst_m,
                                                   amax_m, den_m, xh_mean, b_mean,
                                                   mean_agg, N);
  k_fc16          <<<cdiv(N * 16, 256), 256, 0, stream>>>(mean_agg, fm_W1, fm_b1, t_mean, N);
  k_fc16          <<<cdiv(N * 16, 256), 256, 0, stream>>>(t_mean, fm_W2, fm_b2, mean_out, N);

  // -------- cov path (WMMA GEMMs) --------
  // xh_cov = cov @ W_cov  (50000x256 @ 256x1024), output bf16
  k_swizzle_W<<<cdiv(64 * 8 * 32, 256), 256, 0, stream>>>(W_cov, WfragC, 64, 8, 1024);
  k_swizzle_A<<<cdiv(tilesM * 8 * 32, 256), 256, 0, stream>>>(cov, Afrag, tilesM, 256);
  k_wmma_gemm<<<cdiv(tilesM * 64, 8), 256, 0, stream>>>(Afrag, WfragC, nullptr,
                                                        nullptr, xh_cov,
                                                        tilesM, 64, 1024, 8);
  k_att_cov       <<<cdiv(N * 4, 8), 256, 0, stream>>>(xh_cov, as_cov, ad_cov,
                                                       a_src_c, a_dst_c, N);
  k_softmax_stats <<<cdiv(N, 8), 256, 0, stream>>>(rowptr, srcs, a_src_c, a_dst_c,
                                                   amax_c, den_c, N);
  k_aggregate_cov <<<N, 256, 0, stream>>>(rowptr, srcs, a_src_c, a_dst_c,
                                          amax_c, den_c, xh_cov, b_cov, cov_agg);

  // t_cov   = cov_agg @ fcv_W1 + b1
  k_swizzle_W<<<cdiv(16 * 8 * 32, 256), 256, 0, stream>>>(fcv_W1, Wfrag1, 16, 8, 256);
  k_swizzle_A<<<cdiv(tilesM * 8 * 32, 256), 256, 0, stream>>>(cov_agg, Afrag, tilesM, 256);
  k_wmma_gemm<<<cdiv(tilesM * 16, 8), 256, 0, stream>>>(Afrag, Wfrag1, fcv_b1,
                                                        t_cov, nullptr,
                                                        tilesM, 16, 256, 8);
  // cov_out = t_cov @ fcv_W2 + b2
  k_swizzle_W<<<cdiv(16 * 8 * 32, 256), 256, 0, stream>>>(fcv_W2, Wfrag2, 16, 8, 256);
  k_swizzle_A<<<cdiv(tilesM * 8 * 32, 256), 256, 0, stream>>>(t_cov, Afrag, tilesM, 256);
  k_wmma_gemm<<<cdiv(tilesM * 16, 8), 256, 0, stream>>>(Afrag, Wfrag2, fcv_b2,
                                                        cov_out, nullptr,
                                                        tilesM, 16, 256, 8);
}
